// JointSentenceBiLSTM12_81114752352621
// MI455X (gfx1250) — matrix operand order repaired and checked by reading, not compiled, not run on hardware
//
#include <hip/hip_runtime.h>

#define B_   64
#define L_   128
#define E_   300
#define EP_  320      // E padded to multiple of 32 for WMMA K
#define H_   256
#define G4_  1024     // 4H
#define NE_  34
#define NA_  36
#define MR_  8192     // B*L
#define HS2_ 512      // 2H
#define AWK_ 1092     // 4H + (NA-1) + (NE-1)

#if defined(__has_builtin)
#if __has_builtin(__builtin_amdgcn_global_load_async_to_lds_b128) && \
    __has_builtin(__builtin_amdgcn_s_wait_asynccnt)
#define HAVE_ASYNC_LDS 1
#endif
#endif

typedef __attribute__((ext_vector_type(16))) _Float16 v16h;
typedef __attribute__((ext_vector_type(8)))  _Float16 v8h;
typedef __attribute__((ext_vector_type(8)))  float    v8f;
typedef __attribute__((ext_vector_type(4)))  int      v4i;

#ifdef HAVE_ASYNC_LDS
typedef __attribute__((address_space(1))) v4i as1_v4i;   // global b128 payload
typedef __attribute__((address_space(3))) v4i as3_v4i;   // LDS b128 payload
#endif

__device__ __forceinline__ v16h cat16(v8h lo, v8h hi) {
  return __builtin_shufflevector(lo, hi, 0,1,2,3,4,5,6,7,8,9,10,11,12,13,14,15);
}
__device__ __forceinline__ v8f wmma32f16(v16h a, v16h b, v8f c) {
  // v_wmma_f32_16x16x32_f16: (neg_a, A, neg_b, B, c_mod, C, reuse_a, reuse_b)
  return __builtin_amdgcn_wmma_f32_16x16x32_f16(false, a, false, b, (short)0, c, false, false);
}
__device__ __forceinline__ float sigf(float x) { return 1.0f / (1.0f + __expf(-x)); }

// 16-byte chunk: global -> LDS, async if available
__device__ __forceinline__ void stage16(const _Float16* g, _Float16* l) {
#ifdef HAVE_ASYNC_LDS
  __builtin_amdgcn_global_load_async_to_lds_b128((as1_v4i*)(const void*)g,
                                                 (as3_v4i*)(void*)l, 0, 0);
#else
  *(float4*)l = *(const float4*)g;
#endif
}
__device__ __forceinline__ void stage_wait() {
#ifdef HAVE_ASYNC_LDS
  __builtin_amdgcn_s_wait_asynccnt(0);
#endif
}

// gbuf is stored in the native WMMA D-fragment layout:
//   index = ((ntile*4 + mtile)*32 + lane)*8 + r
// with brow = mtile*16 + (lane>>4)*8 + r, col = ntile*16 + (lane&15).
__device__ __forceinline__ int gidx(int brow, int col) {
  int mtile = brow >> 4, hh = (brow >> 3) & 1, r = brow & 7;
  int ntile = col >> 4, lm = col & 15;
  return (((ntile << 2) + mtile) * 32 + (hh << 4) + lm) * 8 + r;
}

// ---------------------------------------------------------------------------
// Embedding gather -> f16, K padded 300 -> 320 with zeros
// ---------------------------------------------------------------------------
__global__ void gather_embed(const int* __restrict__ ids,
                             const float* __restrict__ tab,
                             _Float16* __restrict__ dst) {
  int i = blockIdx.x * blockDim.x + threadIdx.x;
  if (i >= MR_ * EP_) return;
  int r = i / EP_, k = i - r * EP_;
  float v = 0.f;
  if (k < E_) v = tab[(size_t)ids[r] * E_ + k];
  dst[i] = (_Float16)v;
}

// ---------------------------------------------------------------------------
// Generic f32 -> f16 convert with row/col zero padding (weights for WMMA)
// ---------------------------------------------------------------------------
__global__ void cvt_pad(const float* __restrict__ src, _Float16* __restrict__ dst,
                        int srows, int scols, int sstride, int soff,
                        int dcols, int total) {
  int i = blockIdx.x * blockDim.x + threadIdx.x;
  if (i >= total) return;
  int r = i / dcols, c = i - r * dcols;
  float v = 0.f;
  if (r < srows && c < scols) v = src[(size_t)r * sstride + soff + c];
  dst[i] = (_Float16)v;
}

// ---------------------------------------------------------------------------
// Generic WMMA GEMM: C[M,N] = A[M,K] * B[N,K]^T (+bias), f16 in, f32 out.
// Block tile 128(M) x NT*16(N); 8 waves, wave w owns M-subtile w; K staged in
// LDS in 32-wide chunks (async-to-LDS when the toolchain exposes it).
// ---------------------------------------------------------------------------
template <int NT>
__global__ void gemm_f16_wmma(const _Float16* __restrict__ A, int lda,
                              const _Float16* __restrict__ Bm, int ldb,
                              const float* __restrict__ bias,
                              float* __restrict__ C, int ldc,
                              int K, int nvalid) {
  __shared__ __align__(16) _Float16 sA[128 * 32];
  __shared__ __align__(16) _Float16 sB[NT * 16 * 32];
  const int tid = threadIdx.x;
  const int w = tid >> 5, l = tid & 31;
  const int lm = l & 15, hh = l >> 4;
  const int row0 = blockIdx.x * 128;
  const int col0 = blockIdx.y * (NT * 16);

  v8f zero = {};
  v8f acc[NT];
#pragma unroll
  for (int i = 0; i < NT; ++i) acc[i] = zero;

  for (int k0 = 0; k0 < K; k0 += 32) {
    {  // stage A tile 128x32 (32B per thread)
      int row = tid >> 1, seg = tid & 1;
      const _Float16* s = A + (size_t)(row0 + row) * lda + k0 + seg * 16;
      _Float16* d = sA + row * 32 + seg * 16;
      stage16(s, d);
      stage16(s + 8, d + 8);
      __builtin_prefetch(s + 32, 0, 1);
    }
    if (tid < NT * 16 * 2) {  // stage B tile (NT*16)x32
      int row = tid >> 1, seg = tid & 1;
      const _Float16* s = Bm + (size_t)(col0 + row) * ldb + k0 + seg * 16;
      _Float16* d = sB + row * 32 + seg * 16;
      stage16(s, d);
      stage16(s + 8, d + 8);
    }
    stage_wait();
    __syncthreads();
    // A operand: row = wave M-subtile + lm; ISA A layout (16-bit, 16x32)
    const _Float16* ap = sA + (w * 16 + lm) * 32;
    v16h a = cat16(*(const v8h*)(ap + hh * 8), *(const v8h*)(ap + 16 + hh * 8));
#pragma unroll
    for (int nt = 0; nt < NT; ++nt) {
      const _Float16* bp = sB + (nt * 16 + lm) * 32 + hh * 8;
      v16h b = cat16(*(const v8h*)bp, *(const v8h*)(bp + 16));
      acc[nt] = wmma32f16(a, b, acc[nt]);
    }
    __syncthreads();
  }
  // D layout: VGPR r -> row (hh*8 + r), col lm
#pragma unroll
  for (int nt = 0; nt < NT; ++nt) {
    int col = col0 + nt * 16 + lm;
    if (col < nvalid) {
      float bv = bias ? bias[col] : 0.f;
      int mb = row0 + w * 16 + hh * 8;
#pragma unroll
      for (int r = 0; r < 8; ++r)
        C[(size_t)(mb + r) * ldc + col] = acc[nt][r] + bv;
    }
  }
}

// ---------------------------------------------------------------------------
// BiLSTM recurrence. One block per direction (persistent over all 128 steps).
// h lives in LDS (32KB), c in registers. Per step g[64,1024] = h @ w_hh^T via
// WMMA; B fragments double-buffered in registers inside one nt iteration, but
// the nt loop is kept rolled (#pragma unroll 1) so register pressure stays
// under the 128-VGPR cap of a 1024-thread block (no scratch spills).
// ---------------------------------------------------------------------------
__global__ void lstm_step_all(const float* __restrict__ xg0, const float* __restrict__ xg1,
                              const _Float16* __restrict__ whh0, const _Float16* __restrict__ whh1,
                              float* __restrict__ gbuf, _Float16* __restrict__ hs) {
  __shared__ __align__(16) _Float16 sH[B_ * H_];   // 32KB: current hidden state
  const int dir = blockIdx.x;
  const float* xg = dir ? xg1 : xg0;
  const _Float16* whh = dir ? whh1 : whh0;
  float* g = gbuf + (size_t)dir * (B_ * G4_);
  const int dcol = dir * H_;
  const int tid = threadIdx.x;

  float creg[16];
#pragma unroll
  for (int i = 0; i < 16; ++i) {
    creg[i] = 0.f;
    sH[tid + i * 1024] = (_Float16)0.f;
  }
  __syncthreads();

  const int w = tid >> 5, l = tid & 31;
  const int lm = l & 15, hh = l >> 4;
  const int mtile = w & 3;          // 4 M-subtiles of 16 (batch 64)
  const int ntb = (w >> 2) * 8;     // 8 wave-groups x 8 N-tiles = 64 N-tiles
  v8f zero = {};

  for (int s = 0; s < L_; ++s) {
    int t = dir ? (L_ - 1 - s) : s;
#pragma unroll 1
    for (int nt = 0; nt < 8; ++nt) {
      int ntile = ntb + nt;
      int nrow = ntile * 16 + lm;
      const _Float16* bbase = whh + (size_t)nrow * H_ + hh * 8;
      // register double-buffer for B fragments: load kt+1 while WMMA kt runs
      v16h bcur = cat16(*(const v8h*)bbase, *(const v8h*)(bbase + 16));
      v8f acc = zero;
#pragma unroll
      for (int kt = 0; kt < 8; ++kt) {
        v16h bnext;
        if (kt < 7) {
          const _Float16* bp = bbase + (kt + 1) * 32;
          bnext = cat16(*(const v8h*)bp, *(const v8h*)(bp + 16));
        }
        const _Float16* ap = sH + (mtile * 16 + lm) * H_ + kt * 32;   // LDS read
        v16h a = cat16(*(const v8h*)(ap + hh * 8), *(const v8h*)(ap + 16 + hh * 8));
        acc = wmma32f16(a, bcur, acc);
        if (kt < 7) bcur = bnext;
      }
      // store raw fragment contiguously (2 x b128), no transpose here
      *(v8f*)(g + (((ntile << 2) + mtile) * 32 + l) * 8) = acc;
    }
    __threadfence();
    __syncthreads();
    // gates: i,f,g,o at offsets 0,H,2H,3H (PyTorch order); c in registers
#pragma unroll
    for (int i = 0; i < 16; ++i) {
      int e = tid + i * 1024;
      int b = e >> 8, u = e & 255;
      const float* xr = xg + ((size_t)(b * L_ + t)) * G4_;
      float gi = g[gidx(b, u)]           + xr[u];
      float gf = g[gidx(b, H_ + u)]      + xr[H_ + u];
      float gc = g[gidx(b, 2 * H_ + u)]  + xr[2 * H_ + u];
      float go = g[gidx(b, 3 * H_ + u)]  + xr[3 * H_ + u];
      float c = sigf(gf) * creg[i] + sigf(gi) * tanhf(gc);
      float h = sigf(go) * tanhf(c);
      creg[i] = c;
      sH[e] = (_Float16)h;                              // e == b*H + u
      hs[((size_t)(b * L_ + t)) * HS2_ + dcol + u] = (_Float16)h;
    }
    __syncthreads();
  }
}

// ---------------------------------------------------------------------------
// ev_pred = argmax over NE event logits (first max wins, like jnp.argmax)
// ---------------------------------------------------------------------------
__global__ void ev_argmax(const float* __restrict__ ev, int* __restrict__ evp) {
  int i = blockIdx.x * blockDim.x + threadIdx.x;
  if (i >= MR_) return;
  const float* p = ev + (size_t)i * NE_;
  float best = p[0];
  int bi = 0;
  for (int k = 1; k < NE_; ++k) {
    float v = p[k];
    if (v > best) { best = v; bi = k; }
  }
  evp[i] = bi;
}

// ---------------------------------------------------------------------------
// Sequential scan over trigger positions. One thread per (b, j); contrib of
// the monotone 0/1 g-matrices kept in registers and updated incrementally
// (<=1 new weight column per matrix per step).
// ---------------------------------------------------------------------------
__global__ void arg_scan(const float* __restrict__ basebuf,   // [MR, 36] hs-part + arg_b
                         const float* __restrict__ trigbuf,   // [MR, 36] trig-part
                         const int* __restrict__ evp,         // [MR]
                         const float* __restrict__ arg_w,     // [36, 1092] f32 original
                         float* __restrict__ out_arg) {       // [B, L, L, 36]
  __shared__ float Wg[NA_ * (NA_ - 1)];   // arg_w cols 1024..1058
  __shared__ float Wt[NA_ * (NE_ - 1)];   // arg_w cols 1059..1091
  __shared__ float trig_s[2][NA_];
  const int tid = threadIdx.x;
  for (int i = tid; i < NA_ * (NA_ - 1); i += 256) {
    int k = i / (NA_ - 1), a = i - k * (NA_ - 1);
    Wg[i] = arg_w[(size_t)k * AWK_ + 4 * H_ + a];
  }
  for (int i = tid; i < NA_ * (NE_ - 1); i += 256) {
    int k = i / (NE_ - 1), a = i - k * (NE_ - 1);
    Wt[i] = arg_w[(size_t)k * AWK_ + 4 * H_ + (NA_ - 1) + a];
  }
  const int bloc = tid >> 7;                 // which of 2 batch rows in block
  const int b = blockIdx.x * 2 + bloc;
  const int j = tid & 127;
  const int row = b * L_ + j;

  float base[NA_], contrib[NA_];
#pragma unroll
  for (int k = 0; k < NA_; ++k) {
    base[k] = basebuf[(size_t)row * NA_ + k];
    contrib[k] = 0.f;
  }
  unsigned long long ga = 0ull, gt = 0ull;
  __syncthreads();

  for (int i = 0; i < L_; ++i) {
    if (tid < 2 * NA_) {
      int bb = tid / NA_, k = tid - bb * NA_;
      trig_s[bb][k] = trigbuf[((size_t)((blockIdx.x * 2 + bb) * L_ + i)) * NA_ + k];
    }
    __syncthreads();
    const float* tr = trig_s[bloc];
    float* outp = out_arg + ((size_t)(b * L_ + i) * L_ + j) * NA_;
    float best = -3.4e38f;
    int ai = 0;
#pragma unroll
    for (int k = 0; k < NA_; ++k) {
      float v = base[k] + tr[k] + contrib[k];
      outp[k] = v;
      if (v > best) { best = v; ai = k; }
    }
    int ep = evp[b * L_ + i];
    if (ep > 0 && ai > 0) {               // active & pos
      int ac = ai - 1;                    // g_arg column
      if (!((ga >> ac) & 1ull)) {
        ga |= (1ull << ac);
#pragma unroll
        for (int k = 0; k < NA_; ++k) contrib[k] += Wg[k * (NA_ - 1) + ac];
      }
      int ec = ep - 1;                    // g_trg_arg column
      if (!((gt >> ec) & 1ull)) {
        gt |= (1ull << ec);
#pragma unroll
        for (int k = 0; k < NA_; ++k) contrib[k] += Wt[k * (NE_ - 1) + ec];
      }
    }
    __syncthreads();
  }
}

// ---------------------------------------------------------------------------
extern "C" void kernel_launch(void* const* d_in, const int* in_sizes, int n_in,
                              void* d_out, int out_size, void* d_ws, size_t ws_size,
                              hipStream_t stream) {
  (void)in_sizes; (void)n_in; (void)out_size; (void)ws_size;
  const int*   ids  = (const int*)  d_in[0];
  const float* emb  = (const float*)d_in[1];
  const float* wihf = (const float*)d_in[2];
  const float* whhf = (const float*)d_in[3];
  const float* bf   = (const float*)d_in[4];
  const float* wihb = (const float*)d_in[5];
  const float* whhb = (const float*)d_in[6];
  const float* bb   = (const float*)d_in[7];
  const float* evw  = (const float*)d_in[8];
  const float* evb  = (const float*)d_in[9];
  const float* argw = (const float*)d_in[10];
  const float* argb = (const float*)d_in[11];

  float* out_ev  = (float*)d_out;                       // [MR, 34]
  float* out_arg = out_ev + (size_t)MR_ * NE_;          // [B, L, L, 36]

  char* ws = (char*)d_ws;
  size_t off = 0;
  auto alloc = [&](size_t bytes) -> void* {
    void* p = ws + off;
    off = (off + bytes + 255) & ~(size_t)255;
    return p;
  };
  _Float16* embf   = (_Float16*)alloc((size_t)MR_ * EP_ * 2);
  _Float16* wih16f = (_Float16*)alloc((size_t)G4_ * EP_ * 2);
  _Float16* wih16b = (_Float16*)alloc((size_t)G4_ * EP_ * 2);
  _Float16* whh16f = (_Float16*)alloc((size_t)G4_ * H_ * 2);
  _Float16* whh16b = (_Float16*)alloc((size_t)G4_ * H_ * 2);
  _Float16* evw16  = (_Float16*)alloc((size_t)48 * HS2_ * 2);
  _Float16* awhs16 = (_Float16*)alloc((size_t)48 * HS2_ * 2);
  _Float16* awtr16 = (_Float16*)alloc((size_t)48 * HS2_ * 2);
  float*    xg0    = (float*)   alloc((size_t)MR_ * G4_ * 4);
  float*    xg1    = (float*)   alloc((size_t)MR_ * G4_ * 4);
  _Float16* hs16   = (_Float16*)alloc((size_t)MR_ * HS2_ * 2);
  float*    gbuf   = (float*)   alloc((size_t)2 * B_ * G4_ * 4);
  float*    baseb  = (float*)   alloc((size_t)MR_ * NA_ * 4);
  float*    trigb  = (float*)   alloc((size_t)MR_ * NA_ * 4);
  int*      evp    = (int*)     alloc((size_t)MR_ * 4);

  // 1) gather embeddings (f32 -> f16, pad K to 320)
  gather_embed<<<(MR_ * EP_ + 255) / 256, 256, 0, stream>>>(ids, emb, embf);

  // 2) weight conversions (f32 -> f16, padded for WMMA)
  cvt_pad<<<(G4_ * EP_ + 255) / 256, 256, 0, stream>>>(wihf, wih16f, G4_, E_, E_, 0, EP_, G4_ * EP_);
  cvt_pad<<<(G4_ * EP_ + 255) / 256, 256, 0, stream>>>(wihb, wih16b, G4_, E_, E_, 0, EP_, G4_ * EP_);
  cvt_pad<<<(G4_ * H_ + 255) / 256, 256, 0, stream>>>(whhf, whh16f, G4_, H_, H_, 0, H_, G4_ * H_);
  cvt_pad<<<(G4_ * H_ + 255) / 256, 256, 0, stream>>>(whhb, whh16b, G4_, H_, H_, 0, H_, G4_ * H_);
  cvt_pad<<<(48 * HS2_ + 255) / 256, 256, 0, stream>>>(evw, evw16, NE_, HS2_, HS2_, 0, HS2_, 48 * HS2_);
  cvt_pad<<<(48 * HS2_ + 255) / 256, 256, 0, stream>>>(argw, awhs16, NA_, HS2_, AWK_, 0, HS2_, 48 * HS2_);
  cvt_pad<<<(48 * HS2_ + 255) / 256, 256, 0, stream>>>(argw, awtr16, NA_, HS2_, AWK_, HS2_, HS2_, 48 * HS2_);

  // 3) input-gate GEMMs: xg = emb @ w_ih^T + b   [8192 x 320] x [320 x 1024]
  gemm_f16_wmma<4><<<dim3(64, 16), 256, 0, stream>>>(embf, EP_, wih16f, EP_, bf, xg0, G4_, EP_, G4_);
  gemm_f16_wmma<4><<<dim3(64, 16), 256, 0, stream>>>(embf, EP_, wih16b, EP_, bb, xg1, G4_, EP_, G4_);

  // 4) BiLSTM recurrence: one persistent block per direction
  lstm_step_all<<<2, 1024, 0, stream>>>(xg0, xg1, whh16f, whh16b, gbuf, hs16);

  // 5) step-invariant projections of hs: event logits, base arg part, trig arg part
  gemm_f16_wmma<3><<<dim3(64, 1), 256, 0, stream>>>(hs16, HS2_, evw16, HS2_, evb, out_ev, NE_, HS2_, NE_);
  gemm_f16_wmma<3><<<dim3(64, 1), 256, 0, stream>>>(hs16, HS2_, awhs16, HS2_, argb, baseb, NA_, HS2_, NA_);
  gemm_f16_wmma<3><<<dim3(64, 1), 256, 0, stream>>>(hs16, HS2_, awtr16, HS2_, nullptr, trigb, NA_, HS2_, NA_);

  // 6) event argmax, then the incremental sequential scan producing arg_logits
  ev_argmax<<<32, 256, 0, stream>>>(out_ev, evp);
  arg_scan<<<32, 256, 0, stream>>>(baseb, trigb, evp, argw, out_arg);
}
